// FeatureCorrelator_27788438405902
// MI455X (gfx1250) — matrix operand order repaired and checked by reading, not compiled, not run on hardware
//
#include <hip/hip_runtime.h>
#include <hip/hip_bf16.h>

// ---------------- problem constants (match reference) ----------------
#define BATCH 2
#define NPT1  8192
#define NPT2  8192
#define KNN   16
#define DFEAT 64
#define C0    131      // concat input channels
#define C0P   160      // padded to 5 chunks of 32 for WMMA K
#define CH    128      // hidden / output channels
#define QT    8        // queries per block in the WMMA kernel
#define ROWS  128      // QT * KNN

typedef __attribute__((ext_vector_type(16))) __bf16 v16bf;
typedef __attribute__((ext_vector_type(8)))  float  v8f;
typedef __attribute__((ext_vector_type(4)))  unsigned int v4u;

union Frag { v4u q[2]; unsigned int u[8]; v16bf v; };

__device__ __forceinline__ unsigned short f2bf(float f) {
  union { float f; unsigned int u; } x; x.f = f;
  unsigned int r = x.u + 0x7FFFu + ((x.u >> 16) & 1u);   // round-to-nearest-even
  return (unsigned short)(r >> 16);
}
__device__ __forceinline__ float leakyf(float x) { return x > 0.f ? x : 0.1f * x; }
__device__ __forceinline__ float reluf (float x) { return x > 0.f ? x : 0.f; }

// =====================================================================
// Kernel 1: brute-force KNN (query cloud vs data cloud), LDS-tiled.
// pc layout is [B,3,N] channel-major. One thread per query point.
// Best-16 kept in registers via unrolled replace-max (no scratch).
// =====================================================================
__global__ __launch_bounds__(256) void knn_bruteforce(
    const float* __restrict__ qpc, const float* __restrict__ dpc,
    int nq, int nd, int* __restrict__ idx_out) {
  __shared__ float sx[256], sy[256], sz[256];
  const int b  = blockIdx.y;
  const int qi = blockIdx.x * 256 + threadIdx.x;
  const float* qb = qpc + (size_t)b * 3 * nq;
  const float* db = dpc + (size_t)b * 3 * nd;
  const float qx = qb[qi], qy = qb[nq + qi], qz = qb[2 * nq + qi];

  float bd[KNN]; int bi[KNN];
#pragma unroll
  for (int i = 0; i < KNN; ++i) { bd[i] = 3.0e38f; bi[i] = 0; }
  float worst = 3.0e38f; int wslot = 0;

  for (int t0 = 0; t0 < nd; t0 += 256) {
    __syncthreads();
    const int j = t0 + threadIdx.x;
    sx[threadIdx.x] = db[j];
    sy[threadIdx.x] = db[nd + j];
    sz[threadIdx.x] = db[2 * nd + j];
    __syncthreads();
    for (int j2 = 0; j2 < 256; ++j2) {
      const float dx = sx[j2] - qx, dy = sy[j2] - qy, dz = sz[j2] - qz;
      const float d = dx * dx + dy * dy + dz * dz;
      if (d < worst) {
#pragma unroll
        for (int s = 0; s < KNN; ++s) if (s == wslot) { bd[s] = d; bi[s] = t0 + j2; }
        worst = bd[0]; wslot = 0;
#pragma unroll
        for (int s = 1; s < KNN; ++s) if (bd[s] > worst) { worst = bd[s]; wslot = s; }
      }
    }
  }
  int* o = idx_out + ((size_t)b * nq + qi) * KNN;
#pragma unroll
  for (int i = 0; i < KNN; ++i) o[i] = bi[i];   // order irrelevant: K-reduce is a sum
}

// =====================================================================
// Kernel 2: fused gather + MLP(131->128->128, leaky) + weightnet1 +
// K-reduction, GEMMs via v_wmma_f32_16x16x32_bf16.
// Block = 256 threads (8 waves), handles 8 queries x 16 neighbors = 128 rows.
// Wave w owns row-tile w (== query w): K-reduction is a tile-row reduce.
//
// Weights are staged in LDS in *fragment-contiguous* order:
//   [k-chunk][n-tile][lane][12 dwords]   (8 data dwords + 4 pad)
// so a B-fragment is two ds_load_b128 per lane, bank-conflict-free
// (48 B lane stride: starts 12L mod 64 tile all banks for L=0..15).
//
// LDS layout (bytes):
//   [0,43008)         X bf16 [128][168]  (reused as H [128][136] after L0)
//   [43008,104448)    W0 swizzled: 5 chunks * 8 nt * 32 lanes * 48 B
//   [104448,153600)   W1 swizzled: 4 chunks * 8 nt * 32 lanes * 48 B
//   [153600,157696)   h2 (weightnet hidden) f32 [128][8]
//   [157696,158208)   neighbor idx i32 [128]
// =====================================================================
#define OFF_W0  43008
#define OFF_W1  104448
#define OFF_H2  153600
#define OFF_IDX 157696
#define SM_BYTES 158208

__global__ __launch_bounds__(256) void cross_mlp_wmma(
    const float* __restrict__ pc1, const float* __restrict__ pc2,
    const float* __restrict__ f1,  const float* __restrict__ f2,
    const int*   __restrict__ idx1,
    const float* __restrict__ mW0, const float* __restrict__ mb0,
    const float* __restrict__ mW1, const float* __restrict__ mb1,
    const float* __restrict__ wnW0, const float* __restrict__ wnb0,
    const float* __restrict__ wnW1, const float* __restrict__ wnb1,
    const float* __restrict__ wnW2, const float* __restrict__ wnb2,
    float* __restrict__ nf) {
  __shared__ __align__(16) unsigned char smem[SM_BYTES];
  unsigned short* uX   = (unsigned short*)smem;            // [128][168] halves
  unsigned short* uH   = (unsigned short*)smem;            // reused [128][136]
  const v4u*      xq   = (const v4u*)smem;                 // 16B view of X/H
  unsigned int*   w0s  = (unsigned int*)(smem + OFF_W0);
  unsigned int*   w1s  = (unsigned int*)(smem + OFF_W1);
  const v4u*      w0q  = (const v4u*)(smem + OFF_W0);
  const v4u*      w1q  = (const v4u*)(smem + OFF_W1);
  float*          h2s  = (float*)(smem + OFF_H2);          // [128][8]
  int*            sidx = (int*)(smem + OFF_IDX);           // [128]

  const int b   = blockIdx.y;
  const int n0  = blockIdx.x * QT;
  const int tid = threadIdx.x;

  // ---- stage W0 into LDS, swizzled to B-fragment order ----
  for (int i = tid; i < 5 * 8 * 32 * 8; i += 256) {
    const int chunk = i >> 11;          // k-chunk of 32
    const int rem   = i & 2047;
    const int nt    = rem >> 8;
    const int ln    = (rem >> 3) & 31;
    const int j     = i & 7;
    const int n     = nt * 16 + (ln & 15);
    const int k     = (chunk * 16 + (ln >> 4) * 4 + j + (j >> 2) * 4) * 2;
    unsigned int lo = (k     < C0) ? (unsigned int)f2bf(mW0[n * C0 + k])     : 0u;
    unsigned int hi = (k + 1 < C0) ? (unsigned int)f2bf(mW0[n * C0 + k + 1]) : 0u;
    w0s[((chunk * 8 + nt) * 32 + ln) * 12 + j] = lo | (hi << 16);
  }
  // ---- stage W1 likewise (4 chunks, no tail) ----
  for (int i = tid; i < 4 * 8 * 32 * 8; i += 256) {
    const int chunk = i >> 11;
    const int rem   = i & 2047;
    const int nt    = rem >> 8;
    const int ln    = (rem >> 3) & 31;
    const int j     = i & 7;
    const int n     = nt * 16 + (ln & 15);
    const int k     = (chunk * 16 + (ln >> 4) * 4 + j + (j >> 2) * 4) * 2;
    unsigned int lo = (unsigned int)f2bf(mW1[n * CH + k]);
    unsigned int hi = (unsigned int)f2bf(mW1[n * CH + k + 1]);
    w1s[((chunk * 8 + nt) * 32 + ln) * 12 + j] = lo | (hi << 16);
  }

  // ---- per-row: neighbor idx, direction, weightnet1 hidden h2 ----
  if (tid < ROWS) {
    const int q  = tid >> 4;
    const int n  = n0 + q;
    const int id = idx1[((size_t)b * NPT1 + n) * KNN + (tid & 15)];
    sidx[tid] = id;
    const float qx = pc1[((size_t)b * 3 + 0) * NPT1 + n];
    const float qy = pc1[((size_t)b * 3 + 1) * NPT1 + n];
    const float qz = pc1[((size_t)b * 3 + 2) * NPT1 + n];
    const float dx = pc2[((size_t)b * 3 + 0) * NPT2 + id] - qx;
    const float dy = pc2[((size_t)b * 3 + 1) * NPT2 + id] - qy;
    const float dz = pc2[((size_t)b * 3 + 2) * NPT2 + id] - qz;
    uX[tid * 168 + 128] = f2bf(dx);
    uX[tid * 168 + 129] = f2bf(dy);
    uX[tid * 168 + 130] = f2bf(dz);
#pragma unroll
    for (int c = C0; c < C0P; ++c) uX[tid * 168 + c] = 0;
    float h1[8];
#pragma unroll
    for (int j = 0; j < 8; ++j)
      h1[j] = reluf(wnW0[j * 3] * dx + wnW0[j * 3 + 1] * dy + wnW0[j * 3 + 2] * dz + wnb0[j]);
#pragma unroll
    for (int j = 0; j < 8; ++j) {
      float s = wnb1[j];
#pragma unroll
      for (int i2 = 0; i2 < 8; ++i2) s += wnW1[j * 8 + i2] * h1[i2];
      h2s[tid * 8 + j] = reluf(s);
    }
  }
  __syncthreads();

  // ---- gather features: X[row][0:64]=f1(bcast over k), [64:128]=f2(gather) ----
  {
    const int row = tid >> 1;
    const int seg = tid & 1;
    const int nq  = n0 + (row >> 4);
    const int id  = sidx[row];
    for (int d = 0; d < DFEAT; ++d) {
      const float v = seg ? f2[((size_t)b * DFEAT + d) * NPT2 + id]
                          : f1[((size_t)b * DFEAT + d) * NPT1 + nq];
      uX[row * 168 + seg * DFEAT + d] = f2bf(v);
    }
  }
  __syncthreads();

  const int lane = tid & 31;
  const int wid  = tid >> 5;        // row tile == local query id
  const int m    = lane & 15;
  const int hb   = lane >> 4;       // half-wave select (ISA A/B layout)

  v8f acc[8];
#pragma unroll
  for (int nt = 0; nt < 8; ++nt)
#pragma unroll
    for (int r = 0; r < 8; ++r) acc[nt][r] = 0.f;

  // ---- layer 0: X[128x160] x W0^T[160x128] ----
#pragma unroll
  for (int kc = 0; kc < C0P; kc += 32) {
    Frag a;
    const int abase = (((wid * 16 + m) * 84 + (kc >> 1) + hb * 4) >> 2);
    a.q[0] = xq[abase];
    a.q[1] = xq[abase + 2];
#pragma unroll
    for (int nt = 0; nt < 8; ++nt) {
      Frag bf;
      const int fb = (((kc >> 5) * 8 + nt) * 32 + lane) * 3;   // v4u units
      bf.q[0] = w0q[fb];
      bf.q[1] = w0q[fb + 1];
      acc[nt] = __builtin_amdgcn_wmma_f32_16x16x32_bf16(
          false, a.v, false, bf.v, (short)0, acc[nt], false, false);
    }
  }
  __syncthreads();   // all waves done reading uX before aliased H stores

  // bias + leaky, requantize H to bf16 (C layout: VGPR r -> M = r + 8*hb)
#pragma unroll
  for (int nt = 0; nt < 8; ++nt) {
    const int n = nt * 16 + m;
    const float bias = mb0[n];
#pragma unroll
    for (int r = 0; r < 8; ++r) {
      const int row = wid * 16 + hb * 8 + r;
      uH[row * 136 + n] = f2bf(leakyf(acc[nt][r] + bias));
    }
  }
  __syncthreads();

#pragma unroll
  for (int nt = 0; nt < 8; ++nt)
#pragma unroll
    for (int r = 0; r < 8; ++r) acc[nt][r] = 0.f;

  // ---- layer 1: H[128x128] x W1^T[128x128] ----
#pragma unroll
  for (int kc = 0; kc < CH; kc += 32) {
    Frag a;
    const int abase = (((wid * 16 + m) * 68 + (kc >> 1) + hb * 4) >> 2);
    a.q[0] = xq[abase];
    a.q[1] = xq[abase + 2];
#pragma unroll
    for (int nt = 0; nt < 8; ++nt) {
      Frag bf;
      const int fb = (((kc >> 5) * 8 + nt) * 32 + lane) * 3;
      bf.q[0] = w1q[fb];
      bf.q[1] = w1q[fb + 1];
      acc[nt] = __builtin_amdgcn_wmma_f32_16x16x32_bf16(
          false, a.v, false, bf.v, (short)0, acc[nt], false, false);
    }
  }

  // ---- bias + leaky, apply weightnet1, reduce over K (= tile rows) ----
#pragma unroll
  for (int nt = 0; nt < 8; ++nt) {
    const int n = nt * 16 + m;
    const float bias = mb1[n];
    float w2r[8];
#pragma unroll
    for (int j = 0; j < 8; ++j) w2r[j] = wnW2[n * 8 + j];
    const float bw = wnb2[n];
    float part = 0.f;
#pragma unroll
    for (int r = 0; r < 8; ++r) {
      const int row = wid * 16 + hb * 8 + r;
      const float y = leakyf(acc[nt][r] + bias);
      float wv = bw;
#pragma unroll
      for (int j = 0; j < 8; ++j) wv += w2r[j] * h2s[row * 8 + j];
      part += reluf(wv) * y;
    }
    part += __shfl_xor(part, 16, 32);      // combine M and M+8 halves
    if (hb == 0)
      nf[((size_t)b * NPT1 + (n0 + wid)) * CH + n] = part;
  }
}

// =====================================================================
// Kernel 3: self-KNN re-aggregation. Block = 128 threads = channels,
// one query per block. weightnet2 hidden in LDS; nf gather is L2-resident.
// =====================================================================
__global__ __launch_bounds__(128) void self_aggregate(
    const float* __restrict__ pc1, const int* __restrict__ idx2,
    const float* __restrict__ nf,
    const float* __restrict__ wnW0, const float* __restrict__ wnb0,
    const float* __restrict__ wnW1, const float* __restrict__ wnb1,
    const float* __restrict__ wnW2, const float* __restrict__ wnb2,
    float* __restrict__ out) {
  __shared__ float h2s[KNN * 8];
  __shared__ int sid[KNN];
  const int b = blockIdx.y, n = blockIdx.x, tid = threadIdx.x;
  if (tid < KNN) {
    const int id = idx2[((size_t)b * NPT1 + n) * KNN + tid];
    sid[tid] = id;
    const float dx = pc1[((size_t)b * 3 + 0) * NPT1 + id] - pc1[((size_t)b * 3 + 0) * NPT1 + n];
    const float dy = pc1[((size_t)b * 3 + 1) * NPT1 + id] - pc1[((size_t)b * 3 + 1) * NPT1 + n];
    const float dz = pc1[((size_t)b * 3 + 2) * NPT1 + id] - pc1[((size_t)b * 3 + 2) * NPT1 + n];
    float h1[8];
#pragma unroll
    for (int j = 0; j < 8; ++j)
      h1[j] = reluf(wnW0[j * 3] * dx + wnW0[j * 3 + 1] * dy + wnW0[j * 3 + 2] * dz + wnb0[j]);
#pragma unroll
    for (int j = 0; j < 8; ++j) {
      float s = wnb1[j];
#pragma unroll
      for (int i2 = 0; i2 < 8; ++i2) s += wnW1[j * 8 + i2] * h1[i2];
      h2s[tid * 8 + j] = reluf(s);
    }
  }
  __syncthreads();
  const int c = tid;
  float w2r[8];
#pragma unroll
  for (int j = 0; j < 8; ++j) w2r[j] = wnW2[c * 8 + j];
  const float bw = wnb2[c];
  float acc = 0.f;
#pragma unroll
  for (int k = 0; k < KNN; ++k) {
    float wv = bw;
#pragma unroll
    for (int j = 0; j < 8; ++j) wv += w2r[j] * h2s[k * 8 + j];
    acc += reluf(wv) * nf[((size_t)b * NPT1 + sid[k]) * CH + c];
  }
  out[((size_t)b * CH + c) * NPT1 + n] = acc;   // [B, 128, N1]
}

// =====================================================================
extern "C" void kernel_launch(void* const* d_in, const int* in_sizes, int n_in,
                              void* d_out, int out_size, void* d_ws, size_t ws_size,
                              hipStream_t stream) {
  const float* pc1 = (const float*)d_in[0];
  const float* pc2 = (const float*)d_in[1];
  const float* f1  = (const float*)d_in[2];
  const float* f2  = (const float*)d_in[3];
  const float* mW0 = (const float*)d_in[4];
  const float* mb0 = (const float*)d_in[5];
  const float* mW1 = (const float*)d_in[6];
  const float* mb1 = (const float*)d_in[7];
  const float* w1W0 = (const float*)d_in[8];
  const float* w1b0 = (const float*)d_in[9];
  const float* w1W1 = (const float*)d_in[10];
  const float* w1b1 = (const float*)d_in[11];
  const float* w1W2 = (const float*)d_in[12];
  const float* w1b2 = (const float*)d_in[13];
  const float* w2W0 = (const float*)d_in[14];
  const float* w2b0 = (const float*)d_in[15];
  const float* w2W1 = (const float*)d_in[16];
  const float* w2b1 = (const float*)d_in[17];
  const float* w2W2 = (const float*)d_in[18];
  const float* w2b2 = (const float*)d_in[19];

  int*   idx1 = (int*)d_ws;
  int*   idx2 = idx1 + (size_t)BATCH * NPT1 * KNN;
  float* nf   = (float*)(idx2 + (size_t)BATCH * NPT1 * KNN);

  knn_bruteforce<<<dim3(NPT1 / 256, BATCH), 256, 0, stream>>>(pc1, pc2, NPT1, NPT2, idx1);
  knn_bruteforce<<<dim3(NPT1 / 256, BATCH), 256, 0, stream>>>(pc1, pc1, NPT1, NPT1, idx2);
  cross_mlp_wmma<<<dim3(NPT1 / QT, BATCH), 256, 0, stream>>>(
      pc1, pc2, f1, f2, idx1,
      mW0, mb0, mW1, mb1,
      w1W0, w1b0, w1W1, w1b1, w1W2, w1b2, nf);
  self_aggregate<<<dim3(NPT1, BATCH), 128, 0, stream>>>(
      pc1, idx2, nf, w2W0, w2b0, w2W1, w2b1, w2W2, w2b2, (float*)d_out);
}
